// PointTransformer_19576460935992
// MI455X (gfx1250) — matrix operand order repaired and checked by reading, not compiled, not run on hardware
//
#include <hip/hip_runtime.h>
#include <hip/hip_bf16.h>

typedef __attribute__((ext_vector_type(16))) _Float16 v16h;
typedef __attribute__((ext_vector_type(8)))  _Float16 v8h;
typedef __attribute__((ext_vector_type(8)))  float    v8f;

#define KNB 16

// ---------------------------------------------------------------------------
// Weight pre-swizzle: W f32 [K,N] -> Wp f16 laid out [k/32][n][32] so that a
// WMMA lane's 16 B halves (column n, K sub-range 16*hi..16*hi+15) are
// contiguous and 32B aligned -> two global_load_b128 per fragment.
// ---------------------------------------------------------------------------
__global__ void wswz_kernel(const float* __restrict__ W, _Float16* __restrict__ Wp,
                            int K, int N)
{
    long t = (long)blockIdx.x * blockDim.x + threadIdx.x;
    if (t >= (long)K * N) return;
    int k = (int)(t / N), n = (int)(t % N);
    Wp[((size_t)(k >> 5) * N + n) * 32 + (k & 31)] = (_Float16)W[t];
}

// ---------------------------------------------------------------------------
// WMMA GEMM: C[M,N] = epilogue(A[M,K] @ W[K,N]); one wave per 16x64 C tile
// (4 accumulators, A fragment reused across 4 WMMAs). A is f32 (converted to
// f16 in registers with v_cvt_pk); B is pre-swizzled f16.
// epilogue: t = acc + bias[c]; t = t*gamma[c]+beta[c]; t = relu? max(t,0)
// Requirements: M%16==0, N%64==0, K%32==0.
// ---------------------------------------------------------------------------
__global__ __launch_bounds__(256) void gemm_wmma_kernel(
    const float* __restrict__ A, const _Float16* __restrict__ Wp,
    const float* __restrict__ bias, const float* __restrict__ gamma,
    const float* __restrict__ beta, float* __restrict__ C,
    int M, int N, int Kd, int relu)
{
    const int lane = threadIdx.x & 31;
    const int wave = threadIdx.x >> 5;
    const long tilesN = N >> 6;
    const long tile = (long)blockIdx.x * 8 + wave;
    if (tile >= (long)(M >> 4) * tilesN) return;  // wave-uniform: EXEC stays all-1s
    const int m0 = (int)(tile / tilesN) << 4;
    const int n0 = (int)(tile % tilesN) << 6;

    const int r   = lane & 15;   // A row in tile / B,C column offset
    const int hi  = lane >> 4;
    const int akb = hi ? 16 : 0; // A K sub-base (contiguous k = 16*hi + s, matches B)

    v8f acc0 = {}, acc1 = {}, acc2 = {}, acc3 = {};
    const float* arow = A + (size_t)(m0 + r) * Kd + akb;
    // per-lane B base: column (n0 + g*16 + r), K-halves 16*hi..16*hi+15
    const _Float16* bbase = Wp + ((size_t)n0 + r) * 32 + hi * 16;

    for (int kk = 0; kk < Kd; kk += 32) {
        __builtin_prefetch(arow + kk + 64, 0, 1);
        float4 f0 = *(const float4*)(arow + kk);
        float4 f1 = *(const float4*)(arow + kk + 4);
        float4 f2 = *(const float4*)(arow + kk + 8);
        float4 f3 = *(const float4*)(arow + kk + 12);
        float af[16] = {f0.x,f0.y,f0.z,f0.w, f1.x,f1.y,f1.z,f1.w,
                        f2.x,f2.y,f2.z,f2.w, f3.x,f3.y,f3.z,f3.w};
        v16h av;
        #pragma unroll
        for (int h = 0; h < 16; ++h) av[h] = (_Float16)af[h];

        const _Float16* wp = bbase + (size_t)(kk >> 5) * N * 32;
        v8f* accs[4] = {&acc0, &acc1, &acc2, &acc3};
        #pragma unroll
        for (int g = 0; g < 4; ++g) {
            const _Float16* bp = wp + (size_t)g * 16 * 32;
            v8h b0 = ((const v8h*)bp)[0];
            v8h b1 = ((const v8h*)bp)[1];
            v16h bv;
            #pragma unroll
            for (int h = 0; h < 8; ++h) { bv[h] = b0[h]; bv[h + 8] = b1[h]; }
            *accs[g] = __builtin_amdgcn_wmma_f32_16x16x32_f16(
                false, av, false, bv, (short)0, *accs[g], false, false);
        }
    }

    const int rb = hi * 8;
    v8f accv[4] = {acc0, acc1, acc2, acc3};
    #pragma unroll
    for (int g = 0; g < 4; ++g) {
        const int ncol = n0 + g * 16 + r;
        const float bb = bias ? bias[ncol] : 0.f;
        float sc = 1.f, sh = 0.f;
        if (gamma) { sc = gamma[ncol]; sh = beta[ncol]; }
        #pragma unroll
        for (int v = 0; v < 8; ++v) {
            float t = accv[g][v] + bb;
            t = t * sc + sh;
            if (relu) t = fmaxf(t, 0.f);
            C[(size_t)(m0 + rb + v) * N + ncol] = t;
        }
    }
}

// ---------------------------------------------------------------------------
// mlp_in: x is all-ones [R,1] so out[r,c] = relu((W[c]+b[c])*g[c]+beta[c])
// ---------------------------------------------------------------------------
__global__ void mlp_in_kernel(const float* __restrict__ W, const float* __restrict__ b,
                              const float* __restrict__ beta, const float* __restrict__ gamma,
                              float* __restrict__ x, long R, int d)
{
    long t = (long)blockIdx.x * blockDim.x + threadIdx.x;
    if (t >= R * d) return;
    int c = (int)(t % d);
    float v = (W[c] + b[c]) * gamma[c] + beta[c];
    x[t] = fmaxf(v, 0.f);
}

// ---------------------------------------------------------------------------
// Brute-force kNN (k=16), ascending distance, ties -> lower index.
// ---------------------------------------------------------------------------
__global__ void knn_kernel(const float* __restrict__ pq, const float* __restrict__ ps,
                           int* __restrict__ nb, int Bn, int Mq, int Ns, int exclude)
{
    long t = (long)blockIdx.x * blockDim.x + threadIdx.x;
    if (t >= (long)Bn * Mq) return;
    int b = (int)(t / Mq), m = (int)(t % Mq);
    const float* q = pq + ((size_t)b * Mq + m) * 3;
    float qx = q[0], qy = q[1], qz = q[2];
    float bd[KNB]; int bi[KNB];
    #pragma unroll
    for (int i = 0; i < KNB; ++i) { bd[i] = 3.0e38f; bi[i] = 0; }
    const float* sp = ps + (size_t)b * Ns * 3;
    for (int j = 0; j < Ns; ++j) {
        float dx = qx - sp[3*j], dy = qy - sp[3*j+1], dz = qz - sp[3*j+2];
        float d = dx*dx + dy*dy + dz*dz;
        if (exclude && j == m) d += 1e9f;
        if (d < bd[KNB-1]) {
            int p = KNB - 1;
            while (p > 0 && bd[p-1] > d) { bd[p] = bd[p-1]; bi[p] = bi[p-1]; --p; }
            bd[p] = d; bi[p] = j;
        }
    }
    int* o = nb + ((size_t)b * Mq + m) * KNB;
    #pragma unroll
    for (int k = 0; k < KNB; ++k) o[k] = bi[k];
}

// ---------------------------------------------------------------------------
// Farthest point sampling: one block per batch, min-dist array in LDS.
// ---------------------------------------------------------------------------
__global__ void fps_kernel(const float* __restrict__ pos, int* __restrict__ out,
                           int Np, int m)
{
    __shared__ float mind[2048];
    __shared__ float rv[256];
    __shared__ int   ri[256];
    __shared__ int   s_last;
    const int b = blockIdx.x;
    const int tid = threadIdx.x;
    const float* P = pos + (size_t)b * Np * 3;
    for (int j = tid; j < Np; j += blockDim.x) mind[j] = 3.0e38f;
    if (tid == 0) s_last = 0;
    __syncthreads();
    for (int i = 0; i < m; ++i) {
        int last = s_last;
        if (tid == 0) out[(size_t)b * m + i] = last;
        float lx = P[3*last], ly = P[3*last+1], lz = P[3*last+2];
        float best = -1.f; int bidx = 0;
        for (int j = tid; j < Np; j += blockDim.x) {
            float dx = P[3*j]-lx, dy = P[3*j+1]-ly, dz = P[3*j+2]-lz;
            float d = dx*dx + dy*dy + dz*dz;
            float mn = fminf(mind[j], d);
            mind[j] = mn;
            if (mn > best) { best = mn; bidx = j; }
        }
        rv[tid] = best; ri[tid] = bidx;
        __syncthreads();
        for (int s = blockDim.x >> 1; s > 0; s >>= 1) {
            if (tid < s) {
                if (rv[tid+s] > rv[tid] ||
                    (rv[tid+s] == rv[tid] && ri[tid+s] < ri[tid])) {
                    rv[tid] = rv[tid+s]; ri[tid] = ri[tid+s];
                }
            }
            __syncthreads();
        }
        if (tid == 0) s_last = ri[0];
        __syncthreads();
    }
}

__global__ void gather_pos_kernel(const float* __restrict__ pos, const int* __restrict__ idx,
                                  float* __restrict__ outp, int Bn, int Mnew, int Nold)
{
    long t = (long)blockIdx.x * blockDim.x + threadIdx.x;
    if (t >= (long)Bn * Mnew) return;
    int b = (int)(t / Mnew), m = (int)(t % Mnew);
    int j = idx[(size_t)b * Mnew + m];
    const float* s = pos + ((size_t)b * Nold + j) * 3;
    float* o = outp + ((size_t)b * Mnew + m) * 3;
    o[0] = s[0]; o[1] = s[1]; o[2] = s[2];
}

// h [B*Nold, d] -> out[b,m,c] = max_k h[b, nbr[b,m,k], c]
__global__ void gather_max_kernel(const float* __restrict__ h, const int* __restrict__ nbr,
                                  float* __restrict__ out, int Bn, int Mnew, int Nold, int d)
{
    long t = (long)blockIdx.x * blockDim.x + threadIdx.x;
    if (t >= (long)Bn * Mnew * d) return;
    long r = t / d; int c = (int)(t % d);
    int b = (int)(r / Mnew);
    const int* nn = nbr + (size_t)r * KNB;
    float mx = -3.0e38f;
    #pragma unroll
    for (int k = 0; k < KNB; ++k)
        mx = fmaxf(mx, h[((size_t)b * Nold + nn[k]) * d + c]);
    out[t] = mx;
}

// rel = pos_i - pos_j ; outH[e, 0..63] = relu(rel @ W(3x64) + b)
__global__ void pos_mlp1_kernel(const float* __restrict__ pos, const int* __restrict__ nb,
                                const float* __restrict__ W, const float* __restrict__ bias,
                                float* __restrict__ outH, int Bn, int M)
{
    long e = (long)blockIdx.x * blockDim.x + threadIdx.x;
    long E = (long)Bn * M * KNB;
    if (e >= E) return;
    int b = (int)(e / ((long)M * KNB));
    int rem = (int)(e % ((long)M * KNB));
    int m = rem >> 4;
    int j = nb[e];
    const float* pi = pos + ((size_t)b * M + m) * 3;
    const float* pj = pos + ((size_t)b * M + j) * 3;
    float rx = pi[0] - pj[0], ry = pi[1] - pj[1], rz = pi[2] - pj[2];
    float* o = outH + (size_t)e * 64;
    for (int c = 0; c < 64; ++c)
        o[c] = fmaxf(rx * W[c] + ry * W[64 + c] + rz * W[128 + c] + bias[c], 0.f);
}

// edge_in[e,c] = a_dst[b,m,c] - a_src[b,j,c] + delta[e,c]
__global__ void edge_attin_kernel(const float* __restrict__ adst, const float* __restrict__ asrc,
                                  const float* __restrict__ delta, const int* __restrict__ nb,
                                  float* __restrict__ out, int Bn, int M, int d)
{
    long t = (long)blockIdx.x * blockDim.x + threadIdx.x;
    long total = (long)Bn * M * KNB * d;
    if (t >= total) return;
    long e = t / d; int c = (int)(t % d);
    int b = (int)(e / ((long)M * KNB));
    int rem = (int)(e % ((long)M * KNB));
    int m = rem >> 4;
    int j = nb[e];
    out[t] = adst[((size_t)b * M + m) * d + c]
           - asrc[((size_t)b * M + j) * d + c]
           + delta[t];
}

// per-channel softmax over K neighbors + weighted aggregate of (v_j + delta)
__global__ void aggregate_kernel(const float* __restrict__ alpha, const float* __restrict__ v,
                                 const float* __restrict__ delta, const int* __restrict__ nb,
                                 float* __restrict__ y, int Bn, int M, int d)
{
    long t = (long)blockIdx.x * blockDim.x + threadIdx.x;
    if (t >= (long)Bn * M * d) return;
    long r = t / d; int c = (int)(t % d);
    int b = (int)(r / M);
    long ebase = r * KNB;
    float mx = -3.0e38f;
    #pragma unroll
    for (int k = 0; k < KNB; ++k)
        mx = fmaxf(mx, alpha[(ebase + k) * d + c]);
    float s = 0.f, acc = 0.f;
    #pragma unroll
    for (int k = 0; k < KNB; ++k) {
        float p = __expf(alpha[(ebase + k) * d + c] - mx);
        s += p;
        int j = nb[ebase + k];
        acc += p * (v[((size_t)b * M + j) * d + c] + delta[(ebase + k) * d + c]);
    }
    y[t] = acc / s;
}

__global__ void mean_pool_kernel(const float* __restrict__ x, float* __restrict__ out,
                                 int Bn, int M, int d)
{
    long t = (long)blockIdx.x * blockDim.x + threadIdx.x;
    if (t >= (long)Bn * d) return;
    int b = (int)(t / d), c = (int)(t % d);
    float s = 0.f;
    for (int m = 0; m < M; ++m) s += x[((size_t)b * M + m) * d + c];
    out[t] = s / (float)M;
}

// ---------------------------------------------------------------------------
// Host-side orchestration
// ---------------------------------------------------------------------------
struct BlockP {          // raw fp32 param pointers (pytree order)
    const float *Wdst, *Wsrc, *Wv;
    const float *att1W, *att1b, *att2W, *att2b;
    const float *linInW, *linInb, *linOutW, *linOutb;
    const float *pos1W, *pos1b, *pos2W, *pos2b;
};
struct BlockW {          // pre-swizzled f16 weights + fp32 biases
    const _Float16 *Wdst, *Wsrc, *Wv, *att1, *att2, *linIn, *linOut, *pos2;
    const float *att1b, *att2b, *linInb, *linOutb, *pos1W, *pos1b, *pos2b;
};
struct TDP { const float *W, *b, *beta, *gamma; };
struct Scratch { float *x1, *v, *asrc, *adst, *eH, *eDelta, *eEdge; };

static inline unsigned gridFor(long n) { return (unsigned)((n + 255) >> 8); }

static void launch_gemm(const float* A, const _Float16* Wp, const float* bias,
                        const float* gamma, const float* beta, float* C,
                        long M, int N, int K, int relu, hipStream_t s)
{
    long tiles = (M >> 4) * (long)(N >> 6);   // 16x64 tile per wave
    unsigned blocks = (unsigned)((tiles + 7) >> 3);
    gemm_wmma_kernel<<<blocks, 256, 0, s>>>(A, Wp, bias, gamma, beta, C, (int)M, N, K, relu);
}

static void run_block(const BlockW& P, const float* pos, const int* nb, float* x,
                      int Bn, int M, int d, const Scratch& S, hipStream_t s)
{
    const long R = (long)Bn * M;
    const long E = R * KNB;
    // x1 = relu(x @ lin_in)
    launch_gemm(x, P.linIn, P.linInb, nullptr, nullptr, S.x1, R, d, d, 1, s);
    // v / a_src / a_dst
    launch_gemm(S.x1, P.Wv,   nullptr, nullptr, nullptr, S.v,    R, d, d, 0, s);
    launch_gemm(S.x1, P.Wsrc, nullptr, nullptr, nullptr, S.asrc, R, d, d, 0, s);
    launch_gemm(S.x1, P.Wdst, nullptr, nullptr, nullptr, S.adst, R, d, d, 0, s);
    // delta = relu(relu(rel @ pos1) @ pos2)
    pos_mlp1_kernel<<<gridFor(E), 256, 0, s>>>(pos, nb, P.pos1W, P.pos1b, S.eH, Bn, M);
    launch_gemm(S.eH, P.pos2, P.pos2b, nullptr, nullptr, S.eDelta, E, d, 64, 1, s);
    // attention MLP input, then d->64->d
    edge_attin_kernel<<<gridFor(E * d), 256, 0, s>>>(S.adst, S.asrc, S.eDelta, nb, S.eEdge, Bn, M, d);
    launch_gemm(S.eEdge, P.att1, P.att1b, nullptr, nullptr, S.eH, E, 64, d, 1, s);
    launch_gemm(S.eH, P.att2, P.att2b, nullptr, nullptr, S.eEdge, E, d, 64, 1, s);
    // softmax over K + aggregate -> y (reuse x1), then x = relu(y @ lin_out)
    aggregate_kernel<<<gridFor(R * d), 256, 0, s>>>(S.eEdge, S.v, S.eDelta, nb, S.x1, Bn, M, d);
    launch_gemm(S.x1, P.linOut, P.linOutb, nullptr, nullptr, x, R, d, d, 1, s);
}

extern "C" void kernel_launch(void* const* d_in, const int* in_sizes, int n_in,
                              void* d_out, int out_size, void* d_ws, size_t ws_size,
                              hipStream_t stream)
{
    (void)in_sizes; (void)n_in; (void)out_size; (void)ws_size;
    static const int dims[4] = {64, 128, 256, 512};
    const int Bn = 16;

    // ---- unpack inputs (JAX pytree order: dict keys sorted lexicographically)
    int pi = 0;
    auto nextf = [&]() { return (const float*)d_in[pi++]; };
    const float* pos0 = nextf();
    const float* mlpin_W = nextf(); const float* mlpin_b = nextf();
    const float* mlpin_beta = nextf(); const float* mlpin_gamma = nextf();
    const float* out1_W = nextf(); const float* out1_b = nextf();
    const float* out2_W = nextf(); const float* out2_b = nextf();
    auto readBlock = [&](BlockP& B) {
        B.Wdst = nextf(); B.Wsrc = nextf(); B.Wv = nextf();
        B.att1W = nextf(); B.att1b = nextf(); B.att2W = nextf(); B.att2b = nextf();
        B.linInW = nextf(); B.linInb = nextf(); B.linOutW = nextf(); B.linOutb = nextf();
        B.pos1W = nextf(); B.pos1b = nextf(); B.pos2W = nextf(); B.pos2b = nextf();
    };
    BlockP tbP[3], tbInP;
    for (int i = 0; i < 3; ++i) readBlock(tbP[i]);  // "tb" list
    readBlock(tbInP);                               // "tb_in"
    TDP td[3];
    for (int i = 0; i < 3; ++i) {                   // "td": W, b, beta, gamma
        td[i].W = nextf(); td[i].b = nextf(); td[i].beta = nextf(); td[i].gamma = nextf();
    }

    // ---- workspace carve (bump allocator, 256B aligned)
    char* wsb = (char*)d_ws;
    size_t off = 0;
    auto alloc = [&](size_t bytes) -> void* {
        void* p = wsb + off;
        off += (bytes + 255) & ~(size_t)255;
        return p;
    };
    const long Rmax = (long)Bn * 2048;            // 32768
    const long Emax = Rmax * KNB;                 // 524288
    float* x     = (float*)alloc((size_t)Rmax * 128 * 4);
    float* hbuf  = (float*)alloc((size_t)Rmax * 128 * 4);
    Scratch S;
    S.x1     = (float*)alloc((size_t)Rmax * 64 * 4);
    S.v      = (float*)alloc((size_t)Rmax * 64 * 4);
    S.asrc   = (float*)alloc((size_t)Rmax * 64 * 4);
    S.adst   = (float*)alloc((size_t)Rmax * 64 * 4);
    S.eH     = (float*)alloc((size_t)Emax * 64 * 4);
    S.eDelta = (float*)alloc((size_t)Emax * 64 * 4);
    S.eEdge  = (float*)alloc((size_t)Emax * 64 * 4);
    int*   nbBuf  = (int*)alloc((size_t)Emax * 4);
    int*   nb2Buf = (int*)alloc((size_t)Bn * 512 * KNB * 4);
    int*   fpsIdx = (int*)alloc((size_t)Bn * 512 * 4);
    float* posB   = (float*)alloc((size_t)Bn * 512 * 3 * 4);
    float* posC   = (float*)alloc((size_t)Bn * 512 * 3 * 4);
    float* pooled = (float*)alloc((size_t)Bn * 512 * 4);
    float* hid64  = (float*)alloc((size_t)Bn * 64 * 4);

    // ---- pre-convert + pre-swizzle every GEMM weight to f16 (L2-resident)
    auto prep = [&](const float* W, int K, int N) -> const _Float16* {
        _Float16* p = (_Float16*)alloc((size_t)K * N * 2);
        wswz_kernel<<<gridFor((long)K * N), 256, 0, stream>>>(W, p, K, N);
        return p;
    };
    auto prepBlock = [&](const BlockP& P, int d) -> BlockW {
        BlockW W;
        W.Wdst  = prep(P.Wdst, d, d);   W.Wsrc = prep(P.Wsrc, d, d);
        W.Wv    = prep(P.Wv, d, d);
        W.att1  = prep(P.att1W, d, 64); W.att2 = prep(P.att2W, 64, d);
        W.linIn = prep(P.linInW, d, d); W.linOut = prep(P.linOutW, d, d);
        W.pos2  = prep(P.pos2W, 64, d);
        W.att1b = P.att1b; W.att2b = P.att2b;
        W.linInb = P.linInb; W.linOutb = P.linOutb;
        W.pos1W = P.pos1W; W.pos1b = P.pos1b; W.pos2b = P.pos2b;
        return W;
    };
    BlockW tb[3], tb_in;
    for (int i = 0; i < 3; ++i) tb[i] = prepBlock(tbP[i], dims[i + 1]);
    tb_in = prepBlock(tbInP, 64);
    const _Float16* tdW[3];
    for (int i = 0; i < 3; ++i) tdW[i] = prep(td[i].W, dims[i], dims[i + 1]);
    const _Float16* out1p = prep(out1_W, 512, 64);
    const _Float16* out2p = prep(out2_W, 64, 256);

    // ---- mlp_in (x = ones -> per-channel constant) + first block at d=64
    mlp_in_kernel<<<gridFor(Rmax * 64), 256, 0, stream>>>(
        mlpin_W, mlpin_b, mlpin_beta, mlpin_gamma, x, Rmax, 64);
    knn_kernel<<<gridFor((long)Bn * 2048), 256, 0, stream>>>(
        pos0, pos0, nbBuf, Bn, 2048, 2048, 1);
    run_block(tb_in, pos0, nbBuf, x, Bn, 2048, 64, S, stream);

    // ---- 3 stages of transition-down + transformer block
    const float* posCur = pos0;
    float* posBufs[2] = {posB, posC};
    int Nold = 2048;
    for (int i = 0; i < 3; ++i) {
        const int Mnew = Nold / 4;
        const int dOld = dims[i], dNew = dims[i + 1];
        float* posNext = posBufs[i & 1];
        fps_kernel<<<Bn, 256, 0, stream>>>(posCur, fpsIdx, Nold, Mnew);
        gather_pos_kernel<<<gridFor((long)Bn * Mnew), 256, 0, stream>>>(
            posCur, fpsIdx, posNext, Bn, Mnew, Nold);
        knn_kernel<<<gridFor((long)Bn * Mnew), 256, 0, stream>>>(
            posNext, posCur, nb2Buf, Bn, Mnew, Nold, 0);
        launch_gemm(x, tdW[i], td[i].b, td[i].gamma, td[i].beta,
                    hbuf, (long)Bn * Nold, dNew, dOld, 1, stream);
        gather_max_kernel<<<gridFor((long)Bn * Mnew * dNew), 256, 0, stream>>>(
            hbuf, nb2Buf, x, Bn, Mnew, Nold, dNew);
        knn_kernel<<<gridFor((long)Bn * Mnew), 256, 0, stream>>>(
            posNext, posNext, nbBuf, Bn, Mnew, Mnew, 1);
        run_block(tb[i], posNext, nbBuf, x, Bn, Mnew, dNew, S, stream);
        posCur = posNext;
        Nold = Mnew;
    }

    // ---- global mean pool + head (M=16 -> single WMMA row tile)
    mean_pool_kernel<<<gridFor((long)Bn * 512), 256, 0, stream>>>(x, pooled, Bn, Nold, 512);
    launch_gemm(pooled, out1p, out1_b, nullptr, nullptr, hid64, Bn, 64, 512, 1, stream);
    launch_gemm(hid64, out2p, out2_b, nullptr, nullptr, (float*)d_out, Bn, 256, 64, 0, stream);
}
